// SharpenedCosineSimilarity_60559038874242
// MI455X (gfx1250) — compile-verified, matrix operand order checked by reading, not run on hardware
//
#include <hip/hip_runtime.h>
#include <cstdint>
#include <cstddef>

// ---------------------------------------------------------------------------
// Sharpened Cosine Similarity forward for gfx1250 (MI455X), bf16 WMMA implicit
// GEMM: M = Cout = 256, N = B*H*W = 131072 pixels, K = Cin*3*3 = 1152.
// Block tile: M=256 x N=64 (2 rows x 32 cols). 8 waves; each wave computes a
// 2Mx4N grid of 16x16 tiles = 8 WMMAs per k-step, A fragments prefetched.
// ---------------------------------------------------------------------------

typedef __bf16 bf16_t;
typedef bf16_t v16bf __attribute__((ext_vector_type(16)));
typedef float  v8f   __attribute__((ext_vector_type(8)));
typedef unsigned int v8u __attribute__((ext_vector_type(8)));

#define EPS_F 1e-12f

#define B_    32
#define CIN   128
#define COUT  256
#define H_    64
#define W_    64
#define HW    (H_ * W_)
#define KTOT  (CIN * 9)   // 1152
#define NSTEP 36          // 9 taps * (128/32) channel chunks

// workspace layout in dwords
#define APANEL_OFF 0
#define APANEL_DW  147456                      // 9*4*16 blocks * 32 lanes * 8 dwords
#define DINV_OFF   (APANEL_OFF + APANEL_DW)    // 131072 floats
#define SBUF_OFF   (DINV_OFF + 131072)         // 131072 floats
#define WF_OFF     (SBUF_OFF + 131072)         // 256 floats
#define PSQ_OFF    (WF_OFF + 256)              // 256 floats

// dynamic LDS: B panel K=1152 x N=64 bf16 in fragment layout = 144 KiB
#define BP_DWORDS  (NSTEP * 1024)
#define BP_BYTES   (BP_DWORDS * 4)

__device__ __forceinline__ unsigned short f2bf(float f) {
  union { float f; unsigned u; } v; v.f = f;
  unsigned r = v.u + 0x7FFFu + ((v.u >> 16) & 1u);   // round-to-nearest-even
  return (unsigned short)(r >> 16);
}
__device__ __forceinline__ unsigned pack2bf(float lo, float hi) {
  return (unsigned)f2bf(lo) | ((unsigned)f2bf(hi) << 16);
}

__device__ __forceinline__ v8f wmma_bf16(v8u a, v8u b, v8f c) {
  return __builtin_amdgcn_wmma_f32_16x16x32_bf16(
      false, __builtin_bit_cast(v16bf, a),
      false, __builtin_bit_cast(v16bf, b),
      (short)0, c, false, false);
}

// --------------------------- kernel 0a: weight norms ------------------------
__global__ __launch_bounds__(256) void scs_wprep(const float* __restrict__ Wt,
                                                 const float* __restrict__ P,
                                                 const float* __restrict__ Q,
                                                 float* ws) {
  int oc = threadIdx.x;                       // 256 threads, 1 block
  float qs = Q[0] * 0.01f;
  float q_sqr = qs * qs;
  const float* wrow = Wt + (size_t)oc * KTOT;
  float s = 0.f;
  for (int i = 0; i < KTOT; ++i) { float w = wrow[i]; s += w * w; }
  ws[WF_OFF + oc] = 1.0f / (sqrtf(s) + EPS_F + q_sqr);
  float ps = P[oc] * 0.1f;
  ws[PSQ_OFF + oc] = ps * ps;
}

// --------------------- kernel 0b: swizzled bf16 A panel ---------------------
// A-fragment (16x32 bf16, ISA 7.12.2): lane L (half=L>>4, m=L&15) holds 16
// bf16 packed as dwords j=0..7:
//   j<4 : K = 8*half + 2j, 2j+1        j>=4 : K = 16 + 8*half + 2(j-4), +1
// Block index blk = (tap*4 + kchunk)*16 + ocblock ; dword index = blk*256 +
// lane*8 + j  -> each lane reads its fragment as one contiguous 32B chunk.
__global__ __launch_bounds__(256) void scs_apanel(const float* __restrict__ Wt,
                                                  unsigned* wsu,
                                                  const float* wsf) {
  int e = blockIdx.x * 256 + threadIdx.x;     // < 147456
  int j    = e & 7;
  int lane = (e >> 3) & 31;
  int blk  = e >> 8;                          // (t*4+kc)*16 + ob
  int ob = blk & 15, kc = (blk >> 4) & 3, t = blk >> 6;
  int half = lane >> 4, m = lane & 15;
  int k0 = (j < 4) ? (8 * half + 2 * j) : (16 + 8 * half + 2 * (j - 4));
  int oc = ob * 16 + m;
  int ic = kc * 32 + k0;
  float wf = wsf[WF_OFF + oc];
  size_t widx = ((size_t)oc * CIN + ic) * 9 + t;   // weight[oc][ic][kh][kw]
  float w0 = Wt[widx] * wf;                        // K = k0   (ic)
  float w1 = Wt[widx + 9] * wf;                    // K = k0+1 (ic+1)
  wsu[APANEL_OFF + e] = pack2bf(w0, w1);
}

// ----------------------- kernel 2a: per-pixel sum x^2 -----------------------
__global__ __launch_bounds__(256) void scs_ssq(const float* __restrict__ X,
                                               float* ws) {
  int pix = blockIdx.x * 256 + threadIdx.x;   // < 131072
  int w = pix & 63, h = (pix >> 6) & 63, b = pix >> 12;
  const float* xp = X + (((size_t)b * CIN) * H_ + h) * W_ + w;
  float s = 0.f;
  for (int c = 0; c < CIN; ++c) { float v = xp[(size_t)c * HW]; s += v * v; }
  ws[SBUF_OFF + pix] = s;
}

// ------------------ kernel 2b: 3x3 box sum -> 1/(sqrt+q^2) ------------------
__global__ __launch_bounds__(256) void scs_boxinv(const float* __restrict__ Q,
                                                  float* ws) {
  int pix = blockIdx.x * 256 + threadIdx.x;
  int w = pix & 63, h = (pix >> 6) & 63, b = pix >> 12;
  float qs = Q[0] * 0.01f;
  float q_sqr = qs * qs;
  float s = 0.f;
  for (int dh = -1; dh <= 1; ++dh)
    for (int dw = -1; dw <= 1; ++dw) {
      int r = h + dh, c = w + dw;
      if ((unsigned)r < 64u && (unsigned)c < 64u)
        s += ws[SBUF_OFF + ((b * 64 + r) * 64 + c)];
    }
  ws[DINV_OFF + pix] = 1.0f / (sqrtf(s + EPS_F) + q_sqr);
}

// -------------------------- main WMMA conv kernel ---------------------------
// One workgroup (8 waves) per (b, oh0..oh0+1, ow0..ow0+31) 64-pixel tile,
// computing all 256 output channels. Wave w: oc in [32w, 32w+32) x 64 pixels
// = 2x4 grid of 16x16 WMMA tiles.
__global__ __launch_bounds__(256) void scs_main(const float* __restrict__ X,
                                                const unsigned* wsu,
                                                const float* wsf,
                                                float* __restrict__ Out) {
  extern __shared__ __attribute__((aligned(32))) unsigned Bp[];   // 144 KiB

  int tid  = threadIdx.x;
  int lane = tid & 31;
  int wv   = tid >> 5;
  int bid  = blockIdx.x;                       // 2048 blocks
  int ow0  = (bid & 1) << 5;
  int oh0  = ((bid >> 1) & 31) << 1;
  int b    = bid >> 6;

  // ---- phase 1: build bf16 B panel in LDS --------------------------------
  // B-fragment (32x16): lane L (half=L>>4, n=L&15) holds dwords j=0..7:
  //   j<4 : K = 8*half + 2j, 2j+1       j>=4 : K = 16 + 8*half + 2(j-4), +1
  // Fragment (s, nt) at dword offset s*1024 + nt*256 + lane*8.
  // nt = rsel*2 + csel : pixel row oh0+rsel, cols ow0+csel*16 .. +15.
  {
    int fl = tid & 31;                         // loop-invariant per thread
    int j  = (tid >> 5) & 7;
    int half = fl >> 4, n = fl & 15;
    int k0 = (j < 4) ? (8 * half + 2 * j) : (16 + 8 * half + 2 * (j - 4));
#pragma unroll 8
    for (int i = 0; i < 4 * NSTEP; ++i) {      // 144 dword entries / thread
      int nt = i & 3;
      int s  = i >> 2;                         // k-step 0..35
      int t = s >> 2, kc = s & 3;
      int dh = t / 3, dw = t % 3;
      int rsel = nt >> 1, csel = nt & 1;
      int ic  = kc * 32 + k0;
      int row = oh0 + rsel + dh - 1;
      int col = ow0 + csel * 16 + n + dw - 1;
      float x0 = 0.f, x1 = 0.f;
      if ((unsigned)row < 64u && (unsigned)col < 64u) {
        size_t idx = (((size_t)b * CIN + ic) * H_ + row) * W_ + col;
        x0 = X[idx];
        x1 = X[idx + (size_t)HW];              // ic+1, same pixel
      }
      Bp[s * 1024 + nt * 256 + fl * 8 + j] = pack2bf(x0, x1);
    }
  }
  __syncthreads();

  // ---- phase 2: pipelined K-loop of bf16 WMMAs ---------------------------
  v8f zero8 = {0.f, 0.f, 0.f, 0.f, 0.f, 0.f, 0.f, 0.f};
  v8f acc[2][4];
#pragma unroll
  for (int mt = 0; mt < 2; ++mt)
#pragma unroll
    for (int nt = 0; nt < 4; ++nt) acc[mt][nt] = zero8;

  const v8u* Ap = (const v8u*)(wsu + APANEL_OFF);   // one v8u = one fragment/lane

  // preload step-0 A fragments (global/L2: long latency -> prefetch these)
  v8u a0 = Ap[(size_t)(wv * 2) * 32 + lane];
  v8u a1 = Ap[(size_t)(wv * 2 + 1) * 32 + lane];

  for (int s = 0; s < NSTEP; ++s) {
    // B fragments from LDS (short latency; scheduler hoists across steps)
    v8u bf[4];
#pragma unroll
    for (int nt = 0; nt < 4; ++nt)
      bf[nt] = *(const v8u*)&Bp[s * 1024 + nt * 256 + lane * 8];

    // prefetch next step's A fragments under the 8 WMMAs
    int sn = (s + 1 < NSTEP) ? (s + 1) : s;
    int blkn = sn * 16 + wv * 2;
    v8u na0 = Ap[(size_t)blkn * 32 + lane];
    v8u na1 = Ap[(size_t)(blkn + 1) * 32 + lane];

#pragma unroll
    for (int nt = 0; nt < 4; ++nt) acc[0][nt] = wmma_bf16(a0, bf[nt], acc[0][nt]);
#pragma unroll
    for (int nt = 0; nt < 4; ++nt) acc[1][nt] = wmma_bf16(a1, bf[nt], acc[1][nt]);

    a0 = na0; a1 = na1;
  }

  // ---- epilogue: patch-norm divide, sigmoid * |y|^(p^2) ------------------
  // C/D layout: VGPR r, lanes 0-15 -> M=r, N=lane; lanes 16-31 -> M=8+r.
  int half = lane >> 4, n = lane & 15;

#pragma unroll
  for (int nt = 0; nt < 4; ++nt) {
    int rsel = nt >> 1, csel = nt & 1;
    int oh = oh0 + rsel;
    int colb = ow0 + csel * 16 + n;
    float dinv = wsf[DINV_OFF + ((b * 64 + oh) * 64 + colb)];
    size_t outbase = (((size_t)b * COUT) * H_ + oh) * W_ + colb;
#pragma unroll
    for (int mt = 0; mt < 2; ++mt) {
      int ocb = wv * 32 + mt * 16 + half * 8;
      v8f a = acc[mt][nt];
#pragma unroll
      for (int r = 0; r < 8; ++r) {
        int oc = ocb + r;
        float psq = wsf[PSQ_OFF + oc];
        float y   = a[r] * dinv;
        float sg  = 1.0f / (1.0f + __expf(-y));
        float mag = __powf(fabsf(y) + EPS_F, psq);
        Out[outbase + (size_t)oc * HW] = sg * mag;
      }
    }
  }
}

// ---------------------------------------------------------------------------
extern "C" void kernel_launch(void* const* d_in, const int* in_sizes, int n_in,
                              void* d_out, int out_size, void* d_ws, size_t ws_size,
                              hipStream_t stream) {
  const float* x  = (const float*)d_in[0];
  const float* wt = (const float*)d_in[1];
  const float* p  = (const float*)d_in[2];
  const float* q  = (const float*)d_in[3];
  float*    out = (float*)d_out;
  float*    wsf = (float*)d_ws;
  unsigned* wsu = (unsigned*)d_ws;

  scs_wprep <<<1,    256, 0, stream>>>(wt, p, q, wsf);
  scs_apanel<<<576,  256, 0, stream>>>(wt, wsu, wsf);
  scs_ssq   <<<512,  256, 0, stream>>>(x, wsf);
  scs_boxinv<<<512,  256, 0, stream>>>(q, wsf);
  scs_main  <<<2048, 256, BP_BYTES, stream>>>(x, wsu, wsf, out);
}
